// FNO1dLayer_76209899700631
// MI455X (gfx1250) — matrix operand order repaired and checked by reading, not compiled, not run on hardware
//
#include <hip/hip_runtime.h>

#define B_DIM 32
#define L_DIM 8192
#define C_DIM 64
#define M_DIM 64                 // retained modes
#define J_DIM 128                // 2*M_DIM (re/im interleaved)
#define S_SPLIT 8
#define L_PER_SPLIT (L_DIM / S_SPLIT)        // 1024
#define CHUNKS_PER_SPLIT (L_PER_SPLIT / 32)  // 32

typedef __attribute__((ext_vector_type(16))) __bf16 v16bf;
typedef __attribute__((ext_vector_type(8)))  float  v8f;

union Frag {
    v16bf bf;
    unsigned short u[16];
    uint4 q[2];
};

__device__ __forceinline__ unsigned short f2bf(float f) {
    unsigned int u = __float_as_uint(f);
    u += 0x7FFFu + ((u >> 16) & 1u);   // round-to-nearest-even
    return (unsigned short)(u >> 16);
}

// ---------------------------------------------------------------------------
// Kernel 0: build bf16 DFT bases in workspace.
//   fwd  [J_DIM][L_DIM] : row 2k = cos(2pi k l / L), row 2k+1 = -sin(...)
//   invT [L_DIM][J_DIM] : col 2k = a_k cos(...),     col 2k+1 = -a_k sin(...)
//   a_0 = 1/L, a_k = 2/L  (irfft of a spectrum that is zero above mode 63)
// ---------------------------------------------------------------------------
__global__ void fno_basis_kernel(unsigned short* __restrict__ fwd,
                                 unsigned short* __restrict__ invT) {
    int l = blockIdx.x * blockDim.x + threadIdx.x;
    if (l >= L_DIM) return;
    const float w0 = 6.28318530717958647692f / (float)L_DIM;
    for (int k = 0; k < M_DIM; ++k) {
        int m = (k * l) & (L_DIM - 1);          // exact integer phase reduction
        float a = w0 * (float)m;
        float s, c;
        __sincosf(a, &s, &c);
        fwd[(size_t)(2 * k) * L_DIM + l]     = f2bf(c);
        fwd[(size_t)(2 * k + 1) * L_DIM + l] = f2bf(-s);
        float ak = (k == 0) ? (1.0f / (float)L_DIM) : (2.0f / (float)L_DIM);
        invT[(size_t)l * J_DIM + 2 * k]     = f2bf(ak * c);
        invT[(size_t)l * J_DIM + 2 * k + 1] = f2bf(-ak * s);
    }
}

// ---------------------------------------------------------------------------
// Kernel 1: truncated forward DFT as GEMM.
//   Per workgroup: (batch b, L-split s).  D[j, c] = sum_l fwd[j, l] * x[b,l,c]
//   M = j (8 waves x 16 rows), N = c (4 tiles), K = l (32 chunks of 32).
//   x chunk staged to LDS transposed as bf16 so B-fragment loads are
//   contiguous ds_load_b128 pairs.  Partials per split -> deterministic.
// ---------------------------------------------------------------------------
__global__ void __launch_bounds__(256) fno_dft_kernel(
        const float* __restrict__ x, const unsigned short* __restrict__ fwd,
        float* __restrict__ Xp) {
    __shared__ unsigned short xT[C_DIM * 32];   // [c][l_local], 4 KB

    int b = blockIdx.x >> 3;
    int s = blockIdx.x & 7;
    int l0 = s * L_PER_SPLIT;

    int t = threadIdx.x;
    int lane = t & 31;
    int wave = t >> 5;
    int half = lane >> 4;           // 0: lanes 0-15, 1: lanes 16-31
    int lrow = t >> 3;              // 0..31  (l within chunk, for staging)
    int ccol = (t & 7) * 8;         // 0..56  (c base, for staging)

    v8f acc[4];
    #pragma unroll
    for (int ct = 0; ct < 4; ++ct) acc[ct] = (v8f){0,0,0,0,0,0,0,0};

    const int jrow = wave * 16 + (lane & 15);
    const unsigned short* frow = fwd + (size_t)jrow * L_DIM + l0;

    for (int chunk = 0; chunk < CHUNKS_PER_SPLIT; ++chunk) {
        int lc = chunk * 32;
        const float* src = x + ((size_t)b * L_DIM + l0 + lc + lrow) * C_DIM + ccol;
        float4 p0 = *(const float4*)(src);
        float4 p1 = *(const float4*)(src + 4);
        __builtin_prefetch(src + 32 * C_DIM, 0, 1);   // next chunk -> global_prefetch

        // A fragment: 16x32 bf16, K interleaved per ISA table (global, L2-hot)
        Frag A;
        A.q[0] = *(const uint4*)(frow + lc + half * 8);
        A.q[1] = *(const uint4*)(frow + lc + 16 + half * 8);

        __syncthreads();   // previous iteration's LDS reads complete
        {
            float vals[8] = {p0.x, p0.y, p0.z, p0.w, p1.x, p1.y, p1.z, p1.w};
            #pragma unroll
            for (int i = 0; i < 8; ++i)
                xT[(ccol + i) * 32 + lrow] = f2bf(vals[i]);
        }
        __syncthreads();   // staged chunk visible

        // Preload all four B fragments, then issue WMMAs back-to-back.
        Frag Bf[4];
        #pragma unroll
        for (int ct = 0; ct < 4; ++ct) {
            const unsigned short* brow =
                &xT[(ct * 16 + (lane & 15)) * 32 + half * 8];
            Bf[ct].q[0] = *(const uint4*)(brow);
            Bf[ct].q[1] = *(const uint4*)(brow + 16);
        }
        #pragma unroll
        for (int ct = 0; ct < 4; ++ct)
            acc[ct] = __builtin_amdgcn_wmma_f32_16x16x32_bf16(
                false, A.bf, false, Bf[ct].bf, (short)0, acc[ct], false, false);
    }

    float* xpb = Xp + ((size_t)(s * B_DIM + b)) * J_DIM * C_DIM;
    #pragma unroll
    for (int ct = 0; ct < 4; ++ct) {
        int c = ct * 16 + (lane & 15);
        #pragma unroll
        for (int v = 0; v < 8; ++v) {
            int j = wave * 16 + half * 8 + v;
            xpb[(size_t)j * C_DIM + c] = acc[ct][v];
        }
    }
}

// ---------------------------------------------------------------------------
// Kernel 2: per-mode complex channel mixing (tiny; scalar VALU).
//   Y[b,co,k] = sum_ci X[b,ci,k] * W[ci,co,k]   (complex)
//   One block per (b,k); 64 threads; X partial-splits summed here.
// ---------------------------------------------------------------------------
__global__ void __launch_bounds__(64) fno_mix_kernel(
        const float* __restrict__ Xp, const float* __restrict__ wr,
        const float* __restrict__ wi, unsigned short* __restrict__ Yb) {
    __shared__ float xr_s[C_DIM], xi_s[C_DIM];
    int b = blockIdx.x >> 6;
    int k = blockIdx.x & 63;
    int ci = threadIdx.x;

    float xr = 0.f, xi = 0.f;
    #pragma unroll
    for (int s = 0; s < S_SPLIT; ++s) {
        size_t base = (((size_t)(s * B_DIM + b)) * J_DIM + 2 * k) * C_DIM + ci;
        xr += Xp[base];
        xi += Xp[base + C_DIM];
    }
    xr_s[ci] = xr;
    xi_s[ci] = xi;
    __syncthreads();

    int co = threadIdx.x;
    float aR = 0.f, aI = 0.f;
    for (int c2 = 0; c2 < C_DIM; ++c2) {
        float wrv = wr[((size_t)c2 * C_DIM + co) * M_DIM + k];
        float wiv = wi[((size_t)c2 * C_DIM + co) * M_DIM + k];
        float r = xr_s[c2], im = xi_s[c2];
        aR = fmaf(r, wrv, fmaf(-im, wiv, aR));
        aI = fmaf(r, wiv, fmaf(im, wrv, aI));
    }
    unsigned short* yrow = Yb + (size_t)(b * C_DIM + co) * J_DIM;
    yrow[2 * k]     = f2bf(aR);
    yrow[2 * k + 1] = f2bf(aI);
}

// ---------------------------------------------------------------------------
// Kernel 3: fused inverse DFT + pointwise conv + bias + transpose.
//   Per workgroup: (b, 256-wide l-block). Per wave: two 16-wide l tiles,
//   full 64-channel column (4 M-tiles).  K = 128 (spectral) + 64 (pw).
//   out[b,l,c] = sum_j Y[b,c,j]*invT[l,j] + sum_ci pw[c,ci]*x[b,l,ci] + bias[c]
// ---------------------------------------------------------------------------
__global__ void __launch_bounds__(256) fno_out_kernel(
        const float* __restrict__ x, const unsigned short* __restrict__ Yb,
        const unsigned short* __restrict__ invT, const float* __restrict__ pww,
        const float* __restrict__ pwb, float* __restrict__ out) {
    int b = blockIdx.x >> 5;
    int lblock = (blockIdx.x & 31) * 256;
    int t = threadIdx.x;
    int lane = t & 31;
    int wave = t >> 5;
    int half = lane >> 4;

    int ln[2];
    ln[0] = lblock + wave * 32 + (lane & 15);
    ln[1] = ln[0] + 16;

    v8f acc[2][4];
    #pragma unroll
    for (int nt = 0; nt < 2; ++nt)
        #pragma unroll
        for (int mt = 0; mt < 4; ++mt) acc[nt][mt] = (v8f){0,0,0,0,0,0,0,0};

    // ---- spectral inverse: K = j over 4 chunks of 32 ----
    #pragma unroll
    for (int chunk = 0; chunk < 4; ++chunk) {
        Frag Bn[2];
        #pragma unroll
        for (int nt = 0; nt < 2; ++nt) {
            const unsigned short* p =
                invT + (size_t)ln[nt] * J_DIM + chunk * 32 + half * 8;
            Bn[nt].q[0] = *(const uint4*)(p);
            Bn[nt].q[1] = *(const uint4*)(p + 16);
        }
        #pragma unroll
        for (int mt = 0; mt < 4; ++mt) {
            int co = mt * 16 + (lane & 15);
            const unsigned short* yrow =
                Yb + (size_t)(b * C_DIM + co) * J_DIM + chunk * 32;
            Frag A;
            A.q[0] = *(const uint4*)(yrow + half * 8);
            A.q[1] = *(const uint4*)(yrow + 16 + half * 8);
            #pragma unroll
            for (int nt = 0; nt < 2; ++nt)
                acc[nt][mt] = __builtin_amdgcn_wmma_f32_16x16x32_bf16(
                    false, A.bf, false, Bn[nt].bf, (short)0, acc[nt][mt],
                    false, false);
        }
    }

    // ---- pointwise conv: K = ci over 2 chunks of 32 ----
    #pragma unroll
    for (int chunk = 0; chunk < 2; ++chunk) {
        Frag Bn[2];
        #pragma unroll
        for (int nt = 0; nt < 2; ++nt) {
            const float* p =
                x + ((size_t)b * L_DIM + ln[nt]) * C_DIM + chunk * 32 + half * 8;
            float4 f0 = *(const float4*)(p);
            float4 f1 = *(const float4*)(p + 4);
            float4 f2 = *(const float4*)(p + 16);
            float4 f3 = *(const float4*)(p + 20);
            float vals[16] = {f0.x, f0.y, f0.z, f0.w, f1.x, f1.y, f1.z, f1.w,
                              f2.x, f2.y, f2.z, f2.w, f3.x, f3.y, f3.z, f3.w};
            #pragma unroll
            for (int i = 0; i < 16; ++i) Bn[nt].u[i] = f2bf(vals[i]);
        }
        #pragma unroll
        for (int mt = 0; mt < 4; ++mt) {
            int co = mt * 16 + (lane & 15);
            const float* wrow = pww + (size_t)co * C_DIM + chunk * 32;
            float4 a0 = *(const float4*)(wrow + half * 8);
            float4 a1 = *(const float4*)(wrow + half * 8 + 4);
            float4 a2 = *(const float4*)(wrow + 16 + half * 8);
            float4 a3 = *(const float4*)(wrow + 16 + half * 8 + 4);
            float av[16] = {a0.x, a0.y, a0.z, a0.w, a1.x, a1.y, a1.z, a1.w,
                            a2.x, a2.y, a2.z, a2.w, a3.x, a3.y, a3.z, a3.w};
            Frag A;
            #pragma unroll
            for (int i = 0; i < 16; ++i) A.u[i] = f2bf(av[i]);
            #pragma unroll
            for (int nt = 0; nt < 2; ++nt)
                acc[nt][mt] = __builtin_amdgcn_wmma_f32_16x16x32_bf16(
                    false, A.bf, false, Bn[nt].bf, (short)0, acc[nt][mt],
                    false, false);
        }
    }

    // ---- epilogue: + bias, store transposed [B, L, C] ----
    #pragma unroll
    for (int nt = 0; nt < 2; ++nt) {
        size_t obase = ((size_t)b * L_DIM + ln[nt]) * C_DIM;
        #pragma unroll
        for (int mt = 0; mt < 4; ++mt) {
            int c0 = mt * 16 + half * 8;
            float4 b0 = *(const float4*)(pwb + c0);
            float4 b1 = *(const float4*)(pwb + c0 + 4);
            v8f a = acc[nt][mt];
            float4 r0 = {a[0] + b0.x, a[1] + b0.y, a[2] + b0.z, a[3] + b0.w};
            float4 r1 = {a[4] + b1.x, a[5] + b1.y, a[6] + b1.z, a[7] + b1.w};
            *(float4*)(out + obase + c0)     = r0;
            *(float4*)(out + obase + c0 + 4) = r1;
        }
    }
}

// ---------------------------------------------------------------------------
extern "C" void kernel_launch(void* const* d_in, const int* in_sizes, int n_in,
                              void* d_out, int out_size, void* d_ws, size_t ws_size,
                              hipStream_t stream) {
    (void)in_sizes; (void)n_in; (void)out_size; (void)ws_size;
    const float* x   = (const float*)d_in[0];
    const float* wr  = (const float*)d_in[1];
    const float* wi  = (const float*)d_in[2];
    const float* pww = (const float*)d_in[3];
    const float* pwb = (const float*)d_in[4];
    float* out = (float*)d_out;

    char* ws = (char*)d_ws;
    unsigned short* fwd  = (unsigned short*)(ws);                    // 2 MB
    unsigned short* invT = (unsigned short*)(ws + (2u << 20));       // 2 MB
    float*          Xp   = (float*)(ws + (4u << 20));                // 8 MB
    unsigned short* Yb   = (unsigned short*)(ws + (12u << 20));      // 0.5 MB

    fno_basis_kernel<<<L_DIM / 256, 256, 0, stream>>>(fwd, invT);
    fno_dft_kernel<<<B_DIM * S_SPLIT, 256, 0, stream>>>(x, fwd, Xp);
    fno_mix_kernel<<<B_DIM * M_DIM, 64, 0, stream>>>(Xp, wr, wi, Yb);
    fno_out_kernel<<<B_DIM * (L_DIM / 256), 256, 0, stream>>>(
        x, Yb, invT, pww, pwb, out);
}
